// MambaBlock_56650618634255
// MI455X (gfx1250) — compile-verified, matrix operand order checked by reading, not run on hardware
//
#include <hip/hip_runtime.h>
#include <hip/hip_bf16.h>

// ---------------------------------------------------------------------------
// Problem constants (from reference)
// ---------------------------------------------------------------------------
constexpr int   B_  = 2;
constexpr int   L_  = 8192;
constexpr int   D_  = 1024;
constexpr int   H_  = 16;
constexpr int   N_  = 64;
constexpr int   KC_ = 4;          // depthwise conv width
constexpr int   M_  = B_ * L_;    // 16384 GEMM rows
constexpr int   NC_ = 32;         // scan chunks
constexpr int   LC_ = L_ / NC_;   // 256 steps per chunk
constexpr float DT_MIN_ = 1e-4f, DT_MAX_ = 1e-1f, EPS_ = 1e-6f;

// ---------------------------------------------------------------------------
// Vector types (gfx1250, wave32)
// ---------------------------------------------------------------------------
typedef __attribute__((ext_vector_type(16))) __bf16 v16bf;
typedef __attribute__((ext_vector_type(8)))  __bf16 v8bf;
typedef __attribute__((ext_vector_type(8)))  float  v8f;
typedef __attribute__((ext_vector_type(4)))  unsigned int v4u;
typedef __attribute__((ext_vector_type(8)))  int    v8i;
typedef __attribute__((ext_vector_type(4)))  int    v4i;

__device__ __forceinline__ float fast_sigmoid(float x) {
    return 1.0f / (1.0f + __expf(-x));
}

// ---------------------------------------------------------------------------
// Tensor Data Mover: DMA one rows x 32(bf16) tile, row-padded into LDS.
// pad_enable: after every 16 dwords (64B row) insert 4 dwords (16B)
//   -> LDS row stride 80B = 40 bf16, matching fragment addressing below.
// ---------------------------------------------------------------------------
#define TDM_LDS_STRIDE 40   // bf16 elements per padded LDS row

__device__ __forceinline__ void tdm_load_tile(const void* gptr, unsigned lds_off,
                                              int tile_rows, int tensor_rows,
                                              int k_elems)
{
    unsigned long long ga = (unsigned long long)gptr;
    v4u g0;
    g0[0] = 1u;                                   // count=1 (valid), no gather
    g0[1] = lds_off;                              // lds_addr (bytes)
    g0[2] = (unsigned)(ga & 0xFFFFFFFFu);         // global_addr[31:0]
    g0[3] = (unsigned)((ga >> 32) & 0x01FFFFFFu)  // global_addr[56:32]
          | (2u << 30);                           // type = 2 ("image")
    v8i g1;
    // data_size=2B (code 1), pad_enable, pad_interval=16dw (code 3),
    // pad_amount=4dw (code 3); workgroup_mask = 0 (not in cluster)
    g1[0] = (int)((1u << 16) | (1u << 20) | (3u << 22) | (3u << 25));
    g1[1] = (int)(((unsigned)k_elems & 0xFFFFu) << 16);            // tensor_dim0 lo
    g1[2] = (int)((((unsigned)k_elems >> 16) & 0xFFFFu)            // tensor_dim0 hi
          | (((unsigned)tensor_rows & 0xFFFFu) << 16));            // tensor_dim1 lo
    g1[3] = (int)((((unsigned)tensor_rows >> 16) & 0xFFFFu)        // tensor_dim1 hi
          | (32u << 16));                                          // tile_dim0 = 32
    g1[4] = tile_rows;                             // tile_dim1; tile_dim2 = 0
    g1[5] = k_elems;                               // tensor_dim0_stride[31:0]
    g1[6] = 0;                                     // stride hi / dim1_stride lo
    g1[7] = 0;
    v4i z4 = {0, 0, 0, 0};
#if __clang_major__ >= 23
    v8i z8 = {0, 0, 0, 0, 0, 0, 0, 0};
    __builtin_amdgcn_tensor_load_to_lds(g0, g1, z4, z4, z8, 0);
#else
    __builtin_amdgcn_tensor_load_to_lds(g0, g1, z4, z4, 0);
#endif
}

// ---------------------------------------------------------------------------
// f32 -> bf16 cast (weights)
// ---------------------------------------------------------------------------
__global__ __launch_bounds__(256) void cast_bf16_kernel(
    const float* __restrict__ in, __bf16* __restrict__ out, int n)
{
    int i = blockIdx.x * 256 + threadIdx.x;
    if (i < n) out[i] = (__bf16)in[i];
}

// ---------------------------------------------------------------------------
// RMSNorm: one block per row of D_ elements; writes optional f32 and bf16
// ---------------------------------------------------------------------------
__global__ __launch_bounds__(256) void rmsnorm_kernel(
    const float* __restrict__ x, const float* __restrict__ w,
    float* __restrict__ out_f32, __bf16* __restrict__ out_bf16)
{
    __shared__ float red[256];
    size_t row = blockIdx.x;
    const float* xr = x + row * D_;
    float ss = 0.0f;
    #pragma unroll
    for (int i = threadIdx.x; i < D_; i += 256) { float v = xr[i]; ss += v * v; }
    red[threadIdx.x] = ss;
    __syncthreads();
    for (int s = 128; s > 0; s >>= 1) {
        if (threadIdx.x < (unsigned)s) red[threadIdx.x] += red[threadIdx.x + s];
        __syncthreads();
    }
    float rinv = rsqrtf(red[0] * (1.0f / D_) + EPS_);
    #pragma unroll
    for (int i = threadIdx.x; i < D_; i += 256) {
        float v = xr[i] * rinv * w[i];
        if (out_f32) out_f32[row * D_ + i] = v;
        out_bf16[row * D_ + i] = (__bf16)v;
    }
}

// ---------------------------------------------------------------------------
// bf16 WMMA GEMM:  C[M,N] = A[M,K] * W[N,K]^T  (+ bias, epilogue)
//   mode 0: plain      mode 1: sigmoid      mode 2: add residual
// Block tile 128(M) x 128(N), K step 32.  8 waves in a 4(M) x 2(N) grid;
// each wave owns a 32x64 output patch = 8 x v_wmma_f32_16x16x32_bf16 / K step.
// Staging: TDM double-buffered global->LDS DMA (TENSORcnt), row-padded.
// ---------------------------------------------------------------------------
#define BM 128
#define BN 128
#define BK 32
#define TILE_ELEMS (BM * TDM_LDS_STRIDE)   // 5120 bf16 = 10240 B per buffer

__device__ __forceinline__ v16bf load_frag(const __bf16* rowp)
{
    v8bf lo = *(const v8bf*)(rowp);
    v8bf hi = *(const v8bf*)(rowp + 16);
    return __builtin_shufflevector(lo, hi,
        0, 1, 2, 3, 4, 5, 6, 7, 8, 9, 10, 11, 12, 13, 14, 15);
}

__global__ __launch_bounds__(256) void gemm_bf16_wmma_kernel(
    const __bf16* __restrict__ A, const __bf16* __restrict__ W,
    const float* __restrict__ bias, float* __restrict__ C,
    const float* __restrict__ residual,
    int M, int N, int K, int mode)
{
    __shared__ __bf16 lA[2][TILE_ELEMS];
    __shared__ __bf16 lB[2][TILE_ELEMS];

    const int tid  = threadIdx.x;
    const int wave = tid >> 5;
    const int lane = tid & 31;
    const int mw = wave >> 1;          // 0..3 : 32-row strip
    const int nw = wave & 1;           // 0..1 : 64-col strip
    const int m0 = blockIdx.y * BM;
    const int n0 = blockIdx.x * BN;

    const unsigned ldsA = (unsigned)(uintptr_t)&lA[0][0];
    const unsigned ldsB = (unsigned)(uintptr_t)&lB[0][0];

    v8f acc[2][4] = {};

    const int fr = lane & 15;          // fragment row within 16
    const int kb = (lane >> 4) * 8;    // K sub-block base (0 or 8)

    // Prologue: DMA first K tile into buffer 0
    if (wave == 0) {
        tdm_load_tile(A + (size_t)m0 * K, ldsA, BM, M, K);
        tdm_load_tile(W + (size_t)n0 * K, ldsB, BN, N, K);
    }

    for (int k0 = 0; k0 < K; k0 += BK) {
        const int buf = (k0 / BK) & 1;
        if (wave == 0) __builtin_amdgcn_s_wait_tensorcnt(0);
        __syncthreads();                       // publish LDS tile to all waves
        if (wave == 0 && k0 + BK < K) {        // prefetch next tile -> other buf
            tdm_load_tile(A + (size_t)m0 * K + (k0 + BK), ldsA + (buf ^ 1) * (TILE_ELEMS * 2), BM, M, K);
            tdm_load_tile(W + (size_t)n0 * K + (k0 + BK), ldsB + (buf ^ 1) * (TILE_ELEMS * 2), BN, N, K);
        }

        const __bf16* bufA = &lA[buf][0];
        const __bf16* bufB = &lB[buf][0];

        v16bf afrag[2];
        #pragma unroll
        for (int i = 0; i < 2; ++i)
            afrag[i] = load_frag(&bufA[(mw * 32 + i * 16 + fr) * TDM_LDS_STRIDE + kb]);

        #pragma unroll
        for (int t = 0; t < 4; ++t) {
            v16bf bfrag = load_frag(&bufB[(nw * 64 + t * 16 + fr) * TDM_LDS_STRIDE + kb]);
            #pragma unroll
            for (int i = 0; i < 2; ++i)
                acc[i][t] = __builtin_amdgcn_wmma_f32_16x16x32_bf16(
                    false, afrag[i], false, bfrag, (short)0, acc[i][t], false, false);
        }
        __syncthreads();                       // compute done before buf reuse
    }

    // Epilogue: C/D layout -> lane l, vgpr v: row = v + 8*(l>>4), col = l&15
    const int col   = lane & 15;
    const int rbase = (lane >> 4) * 8;
    #pragma unroll
    for (int i = 0; i < 2; ++i) {
        #pragma unroll
        for (int t = 0; t < 4; ++t) {
            #pragma unroll
            for (int v = 0; v < 8; ++v) {
                int m = m0 + mw * 32 + i * 16 + rbase + v;
                int n = n0 + nw * 64 + t * 16 + col;
                float val = acc[i][t][v] + bias[n];
                if (mode == 1) val = fast_sigmoid(val);
                else if (mode == 2) val += residual[(size_t)m * N + n];
                C[(size_t)m * N + n] = val;
            }
        }
    }
}

// ---------------------------------------------------------------------------
// dt projection (16 cols, trivial) fused with sigmoid -> dt -> delta=exp(-dt)
// ---------------------------------------------------------------------------
__global__ __launch_bounds__(256) void dt_delta_kernel(
    const float* __restrict__ xn, const float* __restrict__ dtW,
    const float* __restrict__ dtb, float* __restrict__ delta)
{
    int idx = blockIdx.x * 256 + threadIdx.x;
    if (idx >= M_ * H_) return;
    int m = idx / H_, h = idx % H_;
    const float* xr = xn  + (size_t)m * D_;
    const float* wr = dtW + (size_t)h * D_;
    float acc = 0.0f;
    #pragma unroll 8
    for (int k = 0; k < D_; ++k) acc = fmaf(xr[k], wr[k], acc);
    acc += dtb[h];
    float dt = DT_MIN_ + fast_sigmoid(acc) * (DT_MAX_ - DT_MIN_);
    delta[idx] = __expf(-dt);
}

// ---------------------------------------------------------------------------
// Causal depthwise conv (K=4) + SiLU, emit bf16 operand for gate GEMM
// ---------------------------------------------------------------------------
__global__ __launch_bounds__(256) void conv_silu_kernel(
    const float* __restrict__ xn, const float* __restrict__ cw,
    const float* __restrict__ cb, __bf16* __restrict__ out)
{
    size_t idx = (size_t)blockIdx.x * 256 + threadIdx.x;
    if (idx >= (size_t)M_ * D_) return;
    int d = (int)(idx % D_);
    int l = (int)((idx / D_) % L_);
    int b = (int)(idx / ((size_t)D_ * L_));
    float acc = cb[d];
    #pragma unroll
    for (int k = 0; k < KC_; ++k) {
        int li = l + k - (KC_ - 1);
        if (li >= 0)
            acc = fmaf(cw[d * KC_ + k], xn[((size_t)b * L_ + li) * D_ + d], acc);
    }
    float s = acc * fast_sigmoid(acc);
    out[idx] = (__bf16)s;
}

// ---------------------------------------------------------------------------
// Chunked parallel scan, phase 1: per-chunk affine summaries (A, S)
//   step map: h -> a_l*h + b_l  with a_l = reset ? 0 : delta
// ---------------------------------------------------------------------------
__global__ __launch_bounds__(64) void scan_p1_kernel(
    const float* __restrict__ delta, const float* __restrict__ param,
    const unsigned char* __restrict__ reset,
    float* __restrict__ Ac, float* __restrict__ Sc)
{
    int blk = blockIdx.x;                  // b*H*NC + h*NC + c
    int c = blk % NC_;
    int h = (blk / NC_) % H_;
    int b = blk / (NC_ * H_);
    int n = threadIdx.x;
    float a = 1.0f, s = 0.0f;
    for (int l = c * LC_; l < (c + 1) * LC_; ++l) {
        size_t bl = (size_t)b * L_ + l;
        float d  = delta[bl * H_ + h];
        float al = reset[bl] ? 0.0f : d;
        float bt = param[(bl * H_ + h) * (2 * N_) + n];   // B_t
        s = fmaf(al, s, bt);
        a *= al;
    }
    Sc[(size_t)blk * N_ + n] = s;
    if (n == 0) Ac[blk] = a;
}

// Phase 2: sequential combine over NC_ chunks (tiny)
__global__ __launch_bounds__(64) void scan_p2_kernel(
    const float* __restrict__ state0, const float* __restrict__ Ac,
    const float* __restrict__ Sc, float* __restrict__ Hinit)
{
    int bh = blockIdx.x;                   // b*H + h
    int n  = threadIdx.x;
    float r = state0[(size_t)bh * N_ + n];
    for (int c = 0; c < NC_; ++c) {
        int blk = bh * NC_ + c;
        Hinit[(size_t)blk * N_ + n] = r;
        r = fmaf(Ac[blk], r, Sc[(size_t)blk * N_ + n]);
    }
}

// Phase 3: replay with correct init state; fuse C_t*h, gating, mix; new_state
__global__ __launch_bounds__(64) void scan_p3_kernel(
    const float* __restrict__ delta, const float* __restrict__ param,
    const unsigned char* __restrict__ reset, const float* __restrict__ Hinit,
    const float* __restrict__ gate, const float* __restrict__ xn,
    float* __restrict__ mixed, float* __restrict__ new_state)
{
    int blk = blockIdx.x;
    int c = blk % NC_;
    int h = (blk / NC_) % H_;
    int b = blk / (NC_ * H_);
    int n = threadIdx.x;
    float hv = Hinit[(size_t)blk * N_ + n];
    int dcol = h * N_ + n;
    for (int l = c * LC_; l < (c + 1) * LC_; ++l) {
        size_t bl = (size_t)b * L_ + l;
        float d  = delta[bl * H_ + h];
        float al = reset[bl] ? 0.0f : d;
        const float* pr = param + (bl * H_ + h) * (2 * N_);
        hv = fmaf(al, hv, pr[n]);          // B_t
        float ct = pr[N_ + n];             // C_t
        float g  = gate[bl * D_ + dcol];
        mixed[bl * D_ + dcol] = fmaf(g, ct * hv - xn[bl * D_ + dcol],
                                     xn[bl * D_ + dcol]);
    }
    if (c == NC_ - 1) new_state[(size_t)(b * H_ + h) * N_ + n] = hv;
}

// ---------------------------------------------------------------------------
// Workspace layout (bytes)
// ---------------------------------------------------------------------------
constexpr size_t S_ROWS  = (size_t)M_ * D_;              // 16,777,216
constexpr size_t OFF_XNF   = 0;                          // x_norm f32
constexpr size_t OFF_XNB   = OFF_XNF   + S_ROWS * 4;     // x_norm bf16
constexpr size_t OFF_PARAM = OFF_XNB   + S_ROWS * 2;     // pp out f32 [M,2048]
constexpr size_t OFF_DELTA = OFF_PARAM + (size_t)M_ * 2 * H_ * N_ * 4;
constexpr size_t OFF_XCB   = OFF_DELTA + (size_t)M_ * H_ * 4;   // conv bf16
constexpr size_t OFF_GATE  = OFF_XCB   + S_ROWS * 2;     // gate f32
constexpr size_t OFF_MIX   = OFF_GATE  + S_ROWS * 4;     // mixed f32
constexpr size_t OFF_MNB   = OFF_MIX   + S_ROWS * 4;     // mixed_norm bf16
constexpr size_t OFF_AC    = OFF_MNB   + S_ROWS * 2;     // chunk A
constexpr size_t OFF_SC    = OFF_AC    + (size_t)B_ * H_ * NC_ * 4;
constexpr size_t OFF_HI    = OFF_SC    + (size_t)B_ * H_ * NC_ * N_ * 4;
constexpr size_t OFF_PPWB  = OFF_HI    + (size_t)B_ * H_ * NC_ * N_ * 4;
constexpr size_t OFF_GWB   = OFF_PPWB  + (size_t)2 * H_ * N_ * D_ * 2;
constexpr size_t OFF_OWB   = OFF_GWB   + (size_t)D_ * D_ * 2;

extern "C" void kernel_launch(void* const* d_in, const int* in_sizes, int n_in,
                              void* d_out, int out_size, void* d_ws, size_t ws_size,
                              hipStream_t stream)
{
    (void)in_sizes; (void)n_in; (void)out_size; (void)ws_size;

    // Inputs in setup_inputs() order
    const float*         x        = (const float*)d_in[0];
    const float*         state    = (const float*)d_in[1];
    const unsigned char* reset    = (const unsigned char*)d_in[2];
    const float*         norm_w   = (const float*)d_in[3];
    const float*         dt_W     = (const float*)d_in[4];
    const float*         dt_b     = (const float*)d_in[5];
    const float*         pp_W     = (const float*)d_in[6];
    const float*         pp_b     = (const float*)d_in[7];
    const float*         conv_w   = (const float*)d_in[8];
    const float*         conv_b   = (const float*)d_in[9];
    const float*         gate_W   = (const float*)d_in[10];
    const float*         gate_b   = (const float*)d_in[11];
    const float*         out_W    = (const float*)d_in[12];
    const float*         out_b    = (const float*)d_in[13];
    const float*         x_norm_w2= (const float*)d_in[14];

    char* ws = (char*)d_ws;
    float*  xnorm  = (float*) (ws + OFF_XNF);
    __bf16* xnormb = (__bf16*)(ws + OFF_XNB);
    float*  param  = (float*) (ws + OFF_PARAM);
    float*  delta  = (float*) (ws + OFF_DELTA);
    __bf16* xconvb = (__bf16*)(ws + OFF_XCB);
    float*  gate   = (float*) (ws + OFF_GATE);
    float*  mixed  = (float*) (ws + OFF_MIX);
    __bf16* mnormb = (__bf16*)(ws + OFF_MNB);
    float*  Ac     = (float*) (ws + OFF_AC);
    float*  Sc     = (float*) (ws + OFF_SC);
    float*  Hinit  = (float*) (ws + OFF_HI);
    __bf16* ppWb   = (__bf16*)(ws + OFF_PPWB);
    __bf16* gWb    = (__bf16*)(ws + OFF_GWB);
    __bf16* oWb    = (__bf16*)(ws + OFF_OWB);

    float* y       = (float*)d_out;
    float* d_state = (float*)d_out + (size_t)M_ * D_;

    // 1. Weight casts to bf16
    cast_bf16_kernel<<<(2 * H_ * N_ * D_ + 255) / 256, 256, 0, stream>>>(pp_W,  ppWb, 2 * H_ * N_ * D_);
    cast_bf16_kernel<<<(D_ * D_ + 255) / 256,       256, 0, stream>>>(gate_W, gWb,  D_ * D_);
    cast_bf16_kernel<<<(D_ * D_ + 255) / 256,       256, 0, stream>>>(out_W,  oWb,  D_ * D_);

    // 2. RMSNorm (f32 + bf16 copies)
    rmsnorm_kernel<<<M_, 256, 0, stream>>>(x, norm_w, xnorm, xnormb);

    // 3. pp projection: [16384,1024] x [2048,1024]^T  (WMMA bf16 + TDM)
    gemm_bf16_wmma_kernel<<<dim3((2 * H_ * N_) / BN, M_ / BM), 256, 0, stream>>>(
        xnormb, ppWb, pp_b, param, nullptr, M_, 2 * H_ * N_, D_, 0);

    // 4. dt -> delta
    dt_delta_kernel<<<(M_ * H_ + 255) / 256, 256, 0, stream>>>(xnorm, dt_W, dt_b, delta);

    // 5. depthwise conv + SiLU -> bf16
    conv_silu_kernel<<<(int)((S_ROWS + 255) / 256), 256, 0, stream>>>(xnorm, conv_w, conv_b, xconvb);

    // 6. gate = sigmoid(conv @ gate_W^T + b)
    gemm_bf16_wmma_kernel<<<dim3(D_ / BN, M_ / BM), 256, 0, stream>>>(
        xconvb, gWb, gate_b, gate, nullptr, M_, D_, D_, 1);

    // 7. chunked parallel scan + fused mix
    scan_p1_kernel<<<B_ * H_ * NC_, N_, 0, stream>>>(delta, param, reset, Ac, Sc);
    scan_p2_kernel<<<B_ * H_,       N_, 0, stream>>>(state, Ac, Sc, Hinit);
    scan_p3_kernel<<<B_ * H_ * NC_, N_, 0, stream>>>(delta, param, reset, Hinit,
                                                     gate, xnorm, mixed, d_state);

    // 8. RMSNorm(mixed) -> bf16 operand
    rmsnorm_kernel<<<M_, 256, 0, stream>>>(mixed, x_norm_w2, nullptr, mnormb);

    // 9. y = mixed_norm @ out_W^T + out_b + x  (WMMA bf16, residual epilogue)
    gemm_bf16_wmma_kernel<<<dim3(D_ / BN, M_ / BM), 256, 0, stream>>>(
        mnormb, oWb, out_b, y, x, M_, D_, D_, 2);
}
